// RelGraphConvLayer_22282290332036
// MI455X (gfx1250) — compile-verified, matrix operand order checked by reading, not run on hardware
//
#include <hip/hip_runtime.h>
#include <hip/hip_bf16.h>

#define NNODES 100000
#define NREL   3
#define NEDGE  600000
#define DFEAT  128   // DIN == DOUT == 128

typedef __attribute__((ext_vector_type(16))) _Float16 v16h;
typedef __attribute__((ext_vector_type(8)))  _Float16 v8h;
typedef __attribute__((ext_vector_type(8)))  float    v8f;
typedef __attribute__((ext_vector_type(4)))  float    v4f;

// ---------------------------------------------------------------------------
// Pack the 4 weight matrices (Wr[0..2], Wl) from row-major [128][128] f32 into
// the per-lane B-operand layout for v_wmma_f32_16x16x32_f16, converted to f16.
// Record layout: [(s*8 + nblock)*4 + kchunk][lane 0..31][16 halves].
// Lane n (0..15)  of B holds column n: K = kc*32 + {0..7, 16..23}
// Lane n+16       of B holds column n: K = kc*32 + {8..15, 24..31}
// ---------------------------------------------------------------------------
__global__ __launch_bounds__(256) void packW_kernel(const float* __restrict__ Wr,
                                                    const float* __restrict__ Wl,
                                                    _Float16* __restrict__ packW) {
    int t = blockIdx.x * 256 + threadIdx.x;   // 0 .. 65535
    int idx = t;
    int i    = t & 15;  t >>= 4;
    int lane = t & 31;  t >>= 5;
    int kc   = t & 3;   t >>= 2;
    int nb   = t & 7;   t >>= 3;
    int s    = t;       // 0..3
    int n = nb * 16 + (lane & 15);
    int k = kc * 32 + ((lane >> 4) ? 8 : 0) + (i < 8 ? i : i + 8);
    const float* W = (s < 3) ? (Wr + (size_t)s * DFEAT * DFEAT) : Wl;
    packW[idx] = (_Float16)W[(size_t)k * DFEAT + n];
}

// ---------------------------------------------------------------------------
// WMMA GEMM: dst[row0:row0+16, :] = x[row0:row0+16, :] @ W_s (+ bias)
// One block = 16 rows, 8 waves; wave w owns output columns [16w, 16w+16).
// K = 128 processed as 4 chunks of 32 via v_wmma_f32_16x16x32_f16.
// ---------------------------------------------------------------------------
__global__ __launch_bounds__(256) void gemm16_kernel(const float* __restrict__ x,
                                                     const _Float16* __restrict__ packW,
                                                     int s,
                                                     const float* __restrict__ bias,
                                                     float* __restrict__ dst) {
    __shared__ _Float16 ldsx[16 * DFEAT];   // 4 KB, f16 x-tile
    const int tid  = threadIdx.x;
    const int row0 = blockIdx.x * 16;

    // Stage + convert the 16x128 f32 tile to f16 in LDS (2048 elems, 512 float4)
    const float4* xv = (const float4*)(x + (size_t)row0 * DFEAT);
    for (int i = tid; i < 512; i += 256) {
        float4 v = xv[i];
        int b = i * 4;
        ldsx[b + 0] = (_Float16)v.x;
        ldsx[b + 1] = (_Float16)v.y;
        ldsx[b + 2] = (_Float16)v.z;
        ldsx[b + 3] = (_Float16)v.w;
    }
    __syncthreads();

    const int lane = tid & 31;
    const int wv   = tid >> 5;        // 0..7 -> N-block
    const int m    = lane & 15;
    const int hi   = lane >> 4;       // 0 or 1
    const int col  = wv * 16 + m;

    v8f acc;
    float bc = bias ? bias[col] : 0.0f;   // bias is per-column, same for all rows
    for (int j = 0; j < 8; ++j) acc[j] = bc;

    for (int kc = 0; kc < 4; ++kc) {
        // A operand: lane<16 -> K {kc*32+0..7, +16..23}; lane>=16 -> shift by 8
        int koff = kc * 32 + hi * 8;
        v8h a0 = *(const v8h*)(&ldsx[m * DFEAT + koff]);
        v8h a1 = *(const v8h*)(&ldsx[m * DFEAT + koff + 16]);
        v16h A;
        for (int i = 0; i < 8; ++i) { A[i] = a0[i]; A[i + 8] = a1[i]; }

        // B operand: 32 contiguous bytes per lane from the packed weights
        const _Float16* brec =
            packW + ((((size_t)s * 8 + wv) * 4 + kc) * 32 + lane) * 16;
        v8h b0 = *(const v8h*)(brec);
        v8h b1 = *(const v8h*)(brec + 8);
        v16h B;
        for (int i = 0; i < 8; ++i) { B[i] = b0[i]; B[i + 8] = b1[i]; }

        acc = __builtin_amdgcn_wmma_f32_16x16x32_f16(
            /*neg_a=*/false, A, /*neg_b=*/false, B,
            /*c_mod=*/(short)0, acc, /*reuse_a=*/false, /*reuse_b=*/false);
    }

    // C/D layout: VGPR j -> row (j | j+8 for hi lanes), column = lane&15 (+16*wv)
    float* dp = dst + (size_t)(row0 + hi * 8) * DFEAT + col;
    for (int j = 0; j < 8; ++j) dp[(size_t)j * DFEAT] = acc[j];
}

// ---------------------------------------------------------------------------
__global__ __launch_bounds__(256) void zero_kernel(float* __restrict__ p, int n) {
    int i = blockIdx.x * 256 + threadIdx.x;
    if (i < n) p[i] = 0.0f;
}

__global__ __launch_bounds__(256) void deg_kernel(const int* __restrict__ edge,
                                                  float* __restrict__ deg) {
    int r = blockIdx.y;
    int e = blockIdx.x * 256 + threadIdx.x;
    if (e < NEDGE) {
        int d = edge[((size_t)r * 2 + 1) * NEDGE + e];
        __hip_atomic_fetch_add(&deg[(size_t)r * NNODES + d], 1.0f,
                               __ATOMIC_RELAXED, __HIP_MEMORY_SCOPE_AGENT);
    }
}

__global__ __launch_bounds__(256) void inv_kernel(const float* __restrict__ deg,
                                                  float* __restrict__ inv, int n) {
    int i = blockIdx.x * 256 + threadIdx.x;
    if (i < n) inv[i] = 1.0f / fmaxf(deg[i], 1.0f);
}

// One wave per edge: lane holds 4 features; out[dst] += h[src] * inv[dst]
__global__ __launch_bounds__(256) void scatter_kernel(const float* __restrict__ h,
                                                      const int* __restrict__ src,
                                                      const int* __restrict__ dst,
                                                      const float* __restrict__ inv,
                                                      float* __restrict__ out) {
    int e = blockIdx.x * 8 + (threadIdx.x >> 5);
    if (e >= NEDGE) return;
    int lane = threadIdx.x & 31;
    int s = src[e];
    int d = dst[e];
    float w = inv[d];
    v4f hv = *(const v4f*)(h + (size_t)s * DFEAT + lane * 4);
    float* op = out + (size_t)d * DFEAT + lane * 4;
    __hip_atomic_fetch_add(op + 0, hv[0] * w, __ATOMIC_RELAXED, __HIP_MEMORY_SCOPE_AGENT);
    __hip_atomic_fetch_add(op + 1, hv[1] * w, __ATOMIC_RELAXED, __HIP_MEMORY_SCOPE_AGENT);
    __hip_atomic_fetch_add(op + 2, hv[2] * w, __ATOMIC_RELAXED, __HIP_MEMORY_SCOPE_AGENT);
    __hip_atomic_fetch_add(op + 3, hv[3] * w, __ATOMIC_RELAXED, __HIP_MEMORY_SCOPE_AGENT);
}

__global__ __launch_bounds__(256) void relu_kernel(float* __restrict__ out, int n4) {
    int i = blockIdx.x * 256 + threadIdx.x;
    if (i < n4) {
        v4f* p = (v4f*)out;
        v4f v = p[i];
        v[0] = fmaxf(v[0], 0.0f);
        v[1] = fmaxf(v[1], 0.0f);
        v[2] = fmaxf(v[2], 0.0f);
        v[3] = fmaxf(v[3], 0.0f);
        p[i] = v;
    }
}

// ---------------------------------------------------------------------------
extern "C" void kernel_launch(void* const* d_in, const int* in_sizes, int n_in,
                              void* d_out, int out_size, void* d_ws, size_t ws_size,
                              hipStream_t stream) {
    const float* x    = (const float*)d_in[0];   // [N,128]
    const int*   edge = (const int*)d_in[1];     // [R,2,E]
    const float* Wr   = (const float*)d_in[2];   // [R,128,128]
    const float* Wl   = (const float*)d_in[3];   // [128,128]
    const float* bl   = (const float*)d_in[4];   // [128]
    float* out = (float*)d_out;

    // Workspace layout
    char* ws = (char*)d_ws;
    _Float16* packW = (_Float16*)ws;                                  // 128 KB
    float* deg = (float*)(ws + 4 * 8 * 4 * 32 * 16 * sizeof(_Float16));
    float* inv = deg + (size_t)NREL * NNODES;                         // 1.2 MB
    float* h   = inv + (size_t)NREL * NNODES;                         // 51.2 MB

    // 1) pack weights into WMMA B-operand layout (f16)
    packW_kernel<<<256, 256, 0, stream>>>(Wr, Wl, packW);

    // 2) degrees -> 1/max(deg,1) per relation
    int n3 = NREL * NNODES;
    zero_kernel<<<(n3 + 255) / 256, 256, 0, stream>>>(deg, n3);
    deg_kernel<<<dim3((NEDGE + 255) / 256, NREL), 256, 0, stream>>>(edge, deg);
    inv_kernel<<<(n3 + 255) / 256, 256, 0, stream>>>(deg, inv, n3);

    // 3) self term initializes out completely: out = x @ Wl + bl
    gemm16_kernel<<<NNODES / 16, 256, 0, stream>>>(x, packW, 3, bl, out);

    // 4) per relation: h = x @ Wr[r]; out[dst] += h[src] / deg[dst]
    for (int r = 0; r < NREL; ++r) {
        gemm16_kernel<<<NNODES / 16, 256, 0, stream>>>(x, packW, r, nullptr, h);
        scatter_kernel<<<NEDGE / 8, 256, 0, stream>>>(
            h,
            edge + (size_t)r * 2 * NEDGE,            // src
            edge + ((size_t)r * 2 + 1) * NEDGE,      // dst
            inv + (size_t)r * NNODES,
            out);
    }

    // 5) ReLU
    relu_kernel<<<(NNODES * DFEAT / 4 + 255) / 256, 256, 0, stream>>>(out, NNODES * DFEAT / 4);
}
